// GATEncoder_61022895341673
// MI455X (gfx1250) — matrix operand order repaired and checked by reading, not compiled, not run on hardware
//
#include <hip/hip_runtime.h>

// ---------------------------------------------------------------------------
// GATEncoder for MI455X (gfx1250): all GEMMs via v_wmma_f32_16x16x32_bf16,
// wave32, LDS-staged tiles, fused bias/relu epilogues, LN + gate kernels.
// Dims: B=8 XN=128 ON=64 D=512 H=8 DH=2048 OBJC=2048 L=2.
// All GEMM dims are multiples of 64 (M) / 32 (K) / 64 (N): exact tiling.
// ---------------------------------------------------------------------------

typedef __attribute__((ext_vector_type(16))) __bf16          v16bf;
typedef __attribute__((ext_vector_type(8)))  float           v8f;
typedef __attribute__((ext_vector_type(8)))  unsigned short  u16x8;

union Frag { u16x8 h[2]; v16bf v; };

__device__ inline unsigned short f2bfbits(float f) {
  union { float f; unsigned u; } x; x.f = f;
  unsigned u = x.u + 0x7FFFu + ((x.u >> 16) & 1u);   // round-to-nearest-even
  return (unsigned short)(u >> 16);
}
__device__ inline __bf16 f2bf(float f) {
  unsigned short h = f2bfbits(f);
  return __builtin_bit_cast(__bf16, h);
}
__device__ inline float bf2f(__bf16 b) {
  unsigned short h = __builtin_bit_cast(unsigned short, b);
  union { unsigned u; float f; } x; x.u = ((unsigned)h) << 16;
  return x.f;
}
__device__ inline v8f wmma_bf16(v16bf a, v16bf b, v8f c) {
  return __builtin_amdgcn_wmma_f32_16x16x32_bf16(false, a, false, b,
                                                 (short)0, c, false, false);
}

// ---------------------------------------------------------------------------
// GEMM: C[M,N] = act(A[M,K] @ W[K,N] + bias + res). act: 0=none, 1=relu.
// Block tile 64x64, 8 waves (4 M-strips x 2 N-strips), K step 32.
// ---------------------------------------------------------------------------
__global__ __launch_bounds__(256)
void gemm_bf16_wmma(const float* __restrict__ A, const float* __restrict__ W,
                    const float* __restrict__ bias, const float* __restrict__ res,
                    float* __restrict__ C, int M, int N, int K, int act) {
  __shared__ __align__(16) __bf16 As[64][40];   // [m][k], pad 8 -> 80B stride
  __shared__ __align__(16) __bf16 Bs[64][40];   // [n][k] (transposed W tile)

  const int tid  = threadIdx.x;
  const int lane = tid & 31, wave = tid >> 5;
  const int wm = wave & 3, wn = wave >> 2;
  const int m0 = blockIdx.y * 64, n0 = blockIdx.x * 64;

  const int arow = tid >> 2, acb = (tid & 3) * 8;   // A tile: 64 rows x 32 k
  const int bk   = tid >> 3, bnb = (tid & 7) * 8;   // W tile: 32 k  x 64 n

  const int mrow = lane & 15;
  const int khalf = (lane >> 4) * 8;
  const int kb = (lane >> 4) * 16;
  const int ncol = lane & 15;

  v8f acc0 = {}; v8f acc1 = {};

  for (int kk = 0; kk < K; kk += 32) {
    // stage A 64x32 (f32 -> bf16, 8 elems/thread, one b128 LDS store)
    {
      const float4* ap = (const float4*)(A + (size_t)(m0 + arow) * K + kk + acb);
      float4 a0 = ap[0], a1 = ap[1];
      uint4 pk;
      pk.x = (unsigned)f2bfbits(a0.x) | ((unsigned)f2bfbits(a0.y) << 16);
      pk.y = (unsigned)f2bfbits(a0.z) | ((unsigned)f2bfbits(a0.w) << 16);
      pk.z = (unsigned)f2bfbits(a1.x) | ((unsigned)f2bfbits(a1.y) << 16);
      pk.w = (unsigned)f2bfbits(a1.z) | ((unsigned)f2bfbits(a1.w) << 16);
      *(uint4*)&As[arow][acb] = pk;
    }
    // stage W 32x64 transposed into Bs[n][k]
    {
      const float4* bp = (const float4*)(W + (size_t)(kk + bk) * N + n0 + bnb);
      float4 b0 = bp[0], b1 = bp[1];
      Bs[bnb + 0][bk] = f2bf(b0.x); Bs[bnb + 1][bk] = f2bf(b0.y);
      Bs[bnb + 2][bk] = f2bf(b0.z); Bs[bnb + 3][bk] = f2bf(b0.w);
      Bs[bnb + 4][bk] = f2bf(b1.x); Bs[bnb + 5][bk] = f2bf(b1.y);
      Bs[bnb + 6][bk] = f2bf(b1.z); Bs[bnb + 7][bk] = f2bf(b1.w);
    }
    __syncthreads();

    Frag af, bf0, bf1;
    af.h[0]  = *(const u16x8*)&As[wm * 16 + mrow][kb];        // K kb..+7
    af.h[1]  = *(const u16x8*)&As[wm * 16 + mrow][kb + 8];   // K kb+8..+15
    bf0.h[0] = *(const u16x8*)&Bs[wn * 32 + ncol][kb];
    bf0.h[1] = *(const u16x8*)&Bs[wn * 32 + ncol][kb + 8];
    bf1.h[0] = *(const u16x8*)&Bs[wn * 32 + 16 + ncol][kb];
    bf1.h[1] = *(const u16x8*)&Bs[wn * 32 + 16 + ncol][kb + 8];
    acc0 = wmma_bf16(af.v, bf0.v, acc0);
    acc1 = wmma_bf16(af.v, bf1.v, acc1);
    __syncthreads();
  }

  // epilogue
  const int mbase = m0 + wm * 16 + ((lane >> 4) << 3);
  const int nA = n0 + wn * 32 + ncol;
  const int nB = nA + 16;
  const float bv0 = bias ? bias[nA] : 0.f;
  const float bv1 = bias ? bias[nB] : 0.f;
#pragma unroll
  for (int r = 0; r < 8; ++r) {
    const int m = mbase + r;
    float v0 = acc0[r] + bv0;
    float v1 = acc1[r] + bv1;
    if (res) { v0 += res[(size_t)m * N + nA]; v1 += res[(size_t)m * N + nB]; }
    if (act == 1) { v0 = fmaxf(v0, 0.f); v1 = fmaxf(v1, 0.f); }
    C[(size_t)m * N + nA] = v0;
    C[(size_t)m * N + nB] = v1;
  }
}

// ---------------------------------------------------------------------------
// Attention: one block per (b, h, qblock-of-64). Q,K,V are [B*Ns, 512] with
// head slice at column h*64; dh = 64; scale = 1/sqrt(64). mask is [B,1,Ns].
// ---------------------------------------------------------------------------
__global__ __launch_bounds__(256)
void attn_wmma(const float* __restrict__ Qg, const float* __restrict__ Kg,
               const float* __restrict__ Vg, const float* __restrict__ maskg,
               float* __restrict__ Og, int Ns) {
  __shared__ __align__(16) __bf16 Qs[64][72];
  __shared__ __align__(16) __bf16 Ks[128][72];
  __shared__ __align__(16) __bf16 Vt[64][136];   // [dh][seq]
  __shared__ __align__(16) __bf16 Ps[64][136];   // scores/probs (bf16)

  const int nqb = Ns >> 6;                 // q-blocks of 64
  const int bh = blockIdx.x / nqb;
  const int qb = blockIdx.x - bh * nqb;
  const int b = bh >> 3, h = bh & 7;
  const int q0 = qb * 64;
  const int tid = threadIdx.x, lane = tid & 31, wave = tid >> 5;

  const size_t base = (size_t)b * Ns * 512 + (size_t)h * 64;

  for (int idx = tid; idx < Ns * 64; idx += 256) {
    const int r = idx >> 6, c = idx & 63;
    const float kv = Kg[base + (size_t)r * 512 + c];
    const float vv = Vg[base + (size_t)r * 512 + c];
    Ks[r][c] = f2bf(kv);
    Vt[c][r] = f2bf(vv);
  }
  for (int idx = tid; idx < 64 * 64; idx += 256) {
    const int r = idx >> 6, c = idx & 63;
    Qs[r][c] = f2bf(Qg[base + (size_t)(q0 + r) * 512 + c]);
  }
  __syncthreads();

  const int mrow = lane & 15;
  const int khalf = (lane >> 4) * 8;
  const int kb = (lane >> 4) * 16;
  const int ncol = lane & 15;
  const int ntn = Ns >> 4;

  // S = scale * Q @ K^T, masked
  for (int t = wave; t < 4 * ntn; t += 8) {
    const int tm = (t / ntn) * 16, tn = (t % ntn) * 16;
    v8f acc = {};
#pragma unroll
    for (int kk = 0; kk < 64; kk += 32) {
      Frag af, bf;
      af.h[0] = *(const u16x8*)&Qs[tm + mrow][kk + kb];
      af.h[1] = *(const u16x8*)&Qs[tm + mrow][kk + kb + 8];
      bf.h[0] = *(const u16x8*)&Ks[tn + ncol][kk + kb];
      bf.h[1] = *(const u16x8*)&Ks[tn + ncol][kk + kb + 8];
      acc = wmma_bf16(af.v, bf.v, acc);
    }
    const int n = tn + ncol;
    const float mk = maskg[(size_t)b * Ns + n];
#pragma unroll
    for (int r = 0; r < 8; ++r) {
      const int m = tm + r + ((lane >> 4) << 3);
      float s = acc[r] * 0.125f;
      s = (mk == 0.f) ? -1e9f : s;
      Ps[m][n] = f2bf(s);
    }
  }
  __syncthreads();

  // row softmax (wave32 shuffle reductions)
  for (int r = wave; r < 64; r += 8) {
    float mx = -3.4e38f;
    for (int c = lane; c < Ns; c += 32) mx = fmaxf(mx, bf2f(Ps[r][c]));
#pragma unroll
    for (int off = 16; off; off >>= 1) mx = fmaxf(mx, __shfl_xor(mx, off, 32));
    float sum = 0.f;
    for (int c = lane; c < Ns; c += 32) {
      const float e = __expf(bf2f(Ps[r][c]) - mx);
      sum += e;
      Ps[r][c] = f2bf(e);
    }
#pragma unroll
    for (int off = 16; off; off >>= 1) sum += __shfl_xor(sum, off, 32);
    const float inv = 1.f / sum;
    for (int c = lane; c < Ns; c += 32) Ps[r][c] = f2bf(bf2f(Ps[r][c]) * inv);
  }
  __syncthreads();

  // O = P @ V
  for (int t = wave; t < 16; t += 8) {
    const int tm = (t >> 2) * 16, tn = (t & 3) * 16;
    v8f acc = {};
    for (int kk = 0; kk < Ns; kk += 32) {
      Frag af, bf;
      af.h[0] = *(const u16x8*)&Ps[tm + mrow][kk + kb];
      af.h[1] = *(const u16x8*)&Ps[tm + mrow][kk + kb + 8];
      bf.h[0] = *(const u16x8*)&Vt[tn + ncol][kk + kb];
      bf.h[1] = *(const u16x8*)&Vt[tn + ncol][kk + kb + 8];
      acc = wmma_bf16(af.v, bf.v, acc);
    }
#pragma unroll
    for (int r = 0; r < 8; ++r) {
      const int m = q0 + tm + r + ((lane >> 4) << 3);
      const int n = tn + ncol;
      Og[base + (size_t)m * 512 + n] = acc[r];
    }
  }
}

// ---------------------------------------------------------------------------
// LayerNorm over D=512: out = g * (h - mean) / sqrt(var + eps) + beta,
// h = a + (res ? res : 0). One block per row, 256 threads (2 elems each).
// ---------------------------------------------------------------------------
__global__ __launch_bounds__(256)
void ln_kernel(float* __restrict__ out, const float* __restrict__ a,
               const float* __restrict__ res, const float* __restrict__ g,
               const float* __restrict__ beta) {
  __shared__ float ssum[8], ssq[8];
  const int row = blockIdx.x, tid = threadIdx.x;
  const size_t base = (size_t)row * 512;
  float h0 = a[base + tid];
  float h1 = a[base + 256 + tid];
  if (res) { h0 += res[base + tid]; h1 += res[base + 256 + tid]; }
  float s = h0 + h1, sq = h0 * h0 + h1 * h1;
#pragma unroll
  for (int off = 16; off; off >>= 1) {
    s += __shfl_xor(s, off, 32);
    sq += __shfl_xor(sq, off, 32);
  }
  if ((tid & 31) == 0) { ssum[tid >> 5] = s; ssq[tid >> 5] = sq; }
  __syncthreads();
  float ts = 0.f, tq = 0.f;
#pragma unroll
  for (int i = 0; i < 8; ++i) { ts += ssum[i]; tq += ssq[i]; }
  const float mean = ts * (1.f / 512.f);
  const float var = tq * (1.f / 512.f) - mean * mean;
  const float inv = rsqrtf(var + 1e-6f);
  out[base + tid]       = g[tid]       * (h0 - mean) * inv + beta[tid];
  out[base + 256 + tid] = g[tid + 256] * (h1 - mean) * inv + beta[tid + 256];
}

// ---------------------------------------------------------------------------
// x2o[b,x,e] = sum_o sigmoid(xa[b,x,e] + ob[b,o,e] + bias[e]) * M[b,x,o]
//              * newo[b,o,e].  One block per (b,x); M is block-uniform.
// ---------------------------------------------------------------------------
__global__ __launch_bounds__(256)
void x2o_gate(const float* __restrict__ xa, const float* __restrict__ ob,
              const float* __restrict__ newo, const float* __restrict__ mat,
              const float* __restrict__ bias, float* __restrict__ out) {
  const int bx = blockIdx.x;               // b*128 + x
  const int b = bx >> 7;
  const int tid = threadIdx.x;
  const size_t rbase = (size_t)bx * 512;
  const float xa0 = xa[rbase + tid] + bias[tid];
  const float xa1 = xa[rbase + 256 + tid] + bias[256 + tid];
  float acc0 = 0.f, acc1 = 0.f;
  for (int o = 0; o < 64; ++o) {
    const float m = mat[(size_t)bx * 64 + o];   // uniform across block
    if (m != 0.f) {
      const size_t obase = ((size_t)b * 64 + o) * 512;
      const float g0 = 1.f / (1.f + __expf(-(xa0 + ob[obase + tid])));
      const float g1 = 1.f / (1.f + __expf(-(xa1 + ob[obase + 256 + tid])));
      acc0 += g0 * m * newo[obase + tid];
      acc1 += g1 * m * newo[obase + 256 + tid];
    }
  }
  out[rbase + tid] = acc0;
  out[rbase + 256 + tid] = acc1;
}

// ---------------------------------------------------------------------------
// o2x[b,o,e] = sum_x sigmoid(xa2[b,x,e] + ob2[b,o,e] + bias[e]) * M[b,x,o]
//              * newx[b,x,e].  One block per (b,o).
// ---------------------------------------------------------------------------
__global__ __launch_bounds__(256)
void o2x_gate(const float* __restrict__ xa2, const float* __restrict__ ob2,
              const float* __restrict__ newx, const float* __restrict__ mat,
              const float* __restrict__ bias, float* __restrict__ out) {
  const int bo = blockIdx.x;               // b*64 + o
  const int b = bo >> 6, o = bo & 63;
  const int tid = threadIdx.x;
  const size_t rbase = (size_t)bo * 512;
  const float ob0 = ob2[rbase + tid] + bias[tid];
  const float ob1 = ob2[rbase + 256 + tid] + bias[256 + tid];
  float acc0 = 0.f, acc1 = 0.f;
  for (int x = 0; x < 128; ++x) {
    const float m = mat[((size_t)b * 128 + x) * 64 + o];
    if (m != 0.f) {
      const size_t xbase = ((size_t)b * 128 + x) * 512;
      const float g0 = 1.f / (1.f + __expf(-(xa2[xbase + tid] + ob0)));
      const float g1 = 1.f / (1.f + __expf(-(xa2[xbase + 256 + tid] + ob1)));
      acc0 += g0 * m * newx[xbase + tid];
      acc1 += g1 * m * newx[xbase + 256 + tid];
    }
  }
  out[rbase + tid] = acc0;
  out[rbase + 256 + tid] = acc1;
}

// ---------------------------------------------------------------------------
// Host orchestration
// ---------------------------------------------------------------------------
namespace {
constexpr int Bz = 8, XN = 128, ON = 64, D = 512, DH = 2048, OBJC = 2048, L = 2;
constexpr int MX = Bz * XN;   // 1024
constexpr int MO = Bz * ON;   // 512
constexpr size_t SZX = (size_t)MX * D;   // 524288
constexpr size_t SZO = (size_t)MO * D;   // 262144
constexpr size_t SZH = (size_t)MX * DH;  // 2097152

// flattened param indices (JAX pytree: dicts in sorted-key order)
constexpr int LBASE = 5, LSTRIDE = 38;
enum {
  ATTO_BK = 0, ATTO_BQ, ATTO_WK, ATTO_WQ,
  ATTX_BK, ATTX_BO, ATTX_BQ, ATTX_BV, ATTX_WK, ATTX_WO, ATTX_WQ, ATTX_WV,
  FFNO_B1, FFNO_B2, FFNO_W1, FFNO_W2,
  FFNX_B1, FFNX_B2, FFNX_W1, FFNX_W2,
  LN_FFN_O_B, LN_FFN_O_G, LN_FFN_X_B, LN_FFN_X_G,
  LN_MES_O_B, LN_MES_O_G, LN_MES_X_B, LN_MES_X_G,
  LN_OGATE_B, LN_OGATE_G, LN_XGATE_B, LN_XGATE_G,
  O2X_B, O2X_WA, O2X_WB, X2O_B, X2O_WA, X2O_WB
};
constexpr int T1B = LBASE + L * LSTRIDE + 0;
constexpr int T1W = LBASE + L * LSTRIDE + 1;
constexpr int T2B = LBASE + L * LSTRIDE + 2;
constexpr int T2W = LBASE + L * LSTRIDE + 3;
}  // namespace

static inline void launch_gemm(const float* A, const float* W, const float* bias,
                               const float* res, float* C, int M, int N, int K,
                               int act, hipStream_t s) {
  dim3 grid(N / 64, M / 64);
  gemm_bf16_wmma<<<grid, 256, 0, s>>>(A, W, bias, res, C, M, N, K, act);
}

extern "C" void kernel_launch(void* const* d_in, const int* in_sizes, int n_in,
                              void* d_out, int out_size, void* d_ws, size_t ws_size,
                              hipStream_t stream) {
  (void)in_sizes; (void)n_in; (void)out_size; (void)ws_size;
  auto IN = [&](int i) { return (const float*)d_in[i]; };

  const float* x_in     = IN(0);
  const float* mask     = IN(1);
  const float* objf     = IN(2);
  const float* obj_mask = IN(3);
  const float* matrix   = IN(4);

  // workspace layout (floats); requires ~31 MB
  float* w = (float*)d_ws;
  float* X   = w;               w += SZX;
  float* O   = w;               w += SZO;
  float* NX  = w;               w += SZX;
  float* NO  = w;               w += SZO;
  float* Qb  = w;               w += SZX;
  float* Kb  = w;               w += SZX;
  float* Vb  = w;               w += SZX;
  float* AT  = w;               w += SZX;
  float* XA  = w;               w += SZX;
  float* OB  = w;               w += SZO;
  float* XA2 = w;               w += SZX;
  float* OB2 = w;               w += SZO;
  float* G1  = w;               w += SZX;
  float* G2  = w;               w += SZO;
  float* HX  = w;               w += SZH;

  float* out_x = (float*)d_out;
  float* out_o = (float*)d_out + SZX;

  // x working copy
  hipMemcpyAsync(X, x_in, SZX * sizeof(float), hipMemcpyDeviceToDevice, stream);

  // trans_obj: relu(relu(objf @ t1w + t1b) @ t2w + t2b)
  launch_gemm(objf, IN(T1W), IN(T1B), nullptr, NO, MO, D, OBJC, 1, stream);
  launch_gemm(NO,   IN(T2W), IN(T2B), nullptr, O,  MO, D, D,    1, stream);

  for (int l = 0; l < L; ++l) {
    const int lb = LBASE + l * LSTRIDE;
    auto P = [&](int k) { return IN(lb + k); };

    // ---- x self-attention (q,k,v,o projections) ----
    launch_gemm(X, P(ATTX_WQ), P(ATTX_BQ), nullptr, Qb, MX, D, D, 0, stream);
    launch_gemm(X, P(ATTX_WK), P(ATTX_BK), nullptr, Kb, MX, D, D, 0, stream);
    launch_gemm(X, P(ATTX_WV), P(ATTX_BV), nullptr, Vb, MX, D, D, 0, stream);
    attn_wmma<<<Bz * 8 * (XN / 64), 256, 0, stream>>>(Qb, Kb, Vb, mask, AT, XN);
    launch_gemm(AT, P(ATTX_WO), P(ATTX_BO), nullptr, G1, MX, D, D, 0, stream);
    ln_kernel<<<MX, 256, 0, stream>>>(NX, G1, X, P(LN_MES_X_G), P(LN_MES_X_B));

    // ---- o self-attention (q,k projections only; v = o, no out proj) ----
    launch_gemm(O, P(ATTO_WQ), P(ATTO_BQ), nullptr, Qb, MO, D, D, 0, stream);
    launch_gemm(O, P(ATTO_WK), P(ATTO_BK), nullptr, Kb, MO, D, D, 0, stream);
    attn_wmma<<<Bz * 8 * (ON / 64), 256, 0, stream>>>(Qb, Kb, O, obj_mask, AT, ON);
    ln_kernel<<<MO, 256, 0, stream>>>(NO, AT, O, P(LN_MES_O_G), P(LN_MES_O_B));

    // ---- gate projections ----
    launch_gemm(NX, P(X2O_WA), nullptr, nullptr, XA,  MX, D, D, 0, stream);
    launch_gemm(NO, P(X2O_WB), nullptr, nullptr, OB,  MO, D, D, 0, stream);
    launch_gemm(NX, P(O2X_WA), nullptr, nullptr, XA2, MX, D, D, 0, stream);
    launch_gemm(NO, P(O2X_WB), nullptr, nullptr, OB2, MO, D, D, 0, stream);

    x2o_gate<<<MX, 256, 0, stream>>>(XA, OB, NO, matrix, P(X2O_B), G1);
    o2x_gate<<<MO, 256, 0, stream>>>(XA2, OB2, NX, matrix, P(O2X_B), G2);

    // gated residual LN -> X/O now hold gated newx/newo
    ln_kernel<<<MX, 256, 0, stream>>>(X, G1, NX, P(LN_XGATE_G), P(LN_XGATE_B));
    ln_kernel<<<MO, 256, 0, stream>>>(O, G2, NO, P(LN_OGATE_G), P(LN_OGATE_B));

    // ---- FFN x ----
    launch_gemm(X,  P(FFNX_W1), P(FFNX_B1), nullptr, HX, MX, DH, D, 1, stream);
    launch_gemm(HX, P(FFNX_W2), P(FFNX_B2), nullptr, G1, MX, D, DH, 0, stream);
    float* xdst = (l == L - 1) ? out_x : X;
    ln_kernel<<<MX, 256, 0, stream>>>(xdst, G1, X, P(LN_FFN_X_G), P(LN_FFN_X_B));

    // ---- FFN o ----
    launch_gemm(O,  P(FFNO_W1), P(FFNO_B1), nullptr, HX, MO, DH, D, 1, stream);
    launch_gemm(HX, P(FFNO_W2), P(FFNO_B2), nullptr, G2, MO, D, DH, 0, stream);
    float* odst = (l == L - 1) ? out_o : O;
    ln_kernel<<<MO, 256, 0, stream>>>(odst, G2, O, P(LN_FFN_O_G), P(LN_FFN_O_B));
  }
}
